// ContentBasedSparseAttention_42717744726489
// MI455X (gfx1250) — matrix operand order repaired and checked by reading, not compiled, not run on hardware
//
#include <hip/hip_runtime.h>
#include <hip/hip_bf16.h>

// ---------------------------------------------------------------------------
// ContentBasedSparseAttention forward for gfx1250 (MI455X, wave32, WMMA).
//   B=4, N=2048, C=768, H=12, hd=64, top_k = int(0.2*2048) = 409
//   d_out = [ out (B*N*C fp32) | attn (B*H*N*N fp32) ]
// GEMMs run on V_WMMA_F32_16X16X32_BF16 with fp32 accumulation; tile staging
// uses GLOBAL_LOAD_ASYNC_TO_LDS_B128 (ASYNCcnt) when the toolchain exposes it.
// ---------------------------------------------------------------------------

typedef __attribute__((ext_vector_type(16))) __bf16 v16bf;
typedef __attribute__((ext_vector_type(8)))  __bf16 v8bf;
typedef __attribute__((ext_vector_type(8)))  float  v8f;
typedef __attribute__((ext_vector_type(4)))  int    v4i;

#define B_    4
#define N_    2048
#define C_    768
#define H_    12
#define HD_   64
#define K3_   2304          // 3*C
#define CH_   384           // C/2 (saliency hidden)
#define TOPK_ 409

// ---- CDNA5 async global->LDS staging (probe-guarded) ----------------------
#if __has_builtin(__builtin_amdgcn_global_load_async_to_lds_b128)
#define HAVE_ASYNC_LDS 1
#else
#define HAVE_ASYNC_LDS 0
#endif

__device__ __forceinline__ void stage16(const void* g, void* lds) {
#if HAVE_ASYNC_LDS
    // signature (probe-decoded): (v4i addrspace(1)* src, v4i addrspace(3)* dst,
    //                             imm offset, imm cpol)
    __builtin_amdgcn_global_load_async_to_lds_b128(
        (__attribute__((address_space(1))) v4i*)(void*)g,
        (__attribute__((address_space(3))) v4i*)lds,
        /*imm offset=*/0, /*cpol=*/0);
#else
    *(uint4*)lds = *(const uint4*)g;
#endif
}

__device__ __forceinline__ void stage_wait() {
#if HAVE_ASYNC_LDS
#if __has_builtin(__builtin_amdgcn_s_wait_asynccnt)
    __builtin_amdgcn_s_wait_asynccnt(0);
#else
    asm volatile("s_wait_asynccnt 0x0" ::: "memory");
#endif
#endif
}

// ---------------------------------------------------------------------------
// Saliency MLP:  sal[row] = gelu(x_row @ w1 + b1) @ w2 + b2     (row = b*N+n)
// One block (384 threads == CH_ hidden units) per token row.
// ---------------------------------------------------------------------------
__global__ __launch_bounds__(CH_)
void sal_kernel(const float* __restrict__ x,  const float* __restrict__ w1,
                const float* __restrict__ b1, const float* __restrict__ w2,
                const float* __restrict__ b2, float* __restrict__ sal) {
    __shared__ float xs[C_];
    __shared__ float red[CH_];
    const int row = blockIdx.x;
    const int tid = threadIdx.x;

    for (int i = tid; i < C_; i += CH_) xs[i] = x[(size_t)row * C_ + i];
    __syncthreads();

    float h = b1[tid];
    #pragma unroll 4
    for (int i = 0; i < C_; ++i) h += xs[i] * w1[i * CH_ + tid];
    // exact GELU: 0.5*h*(1+erf(h/sqrt(2)))
    const float g = 0.5f * h * (1.0f + erff(h * 0.70710678118654752f));
    red[tid] = g * w2[tid];
    __syncthreads();
    if (tid < 128) red[tid] += red[tid + 256];     // fold 384 -> 256
    __syncthreads();
    for (int s = 128; s > 0; s >>= 1) {
        if (tid < s) red[tid] += red[tid + s];
        __syncthreads();
    }
    if (tid == 0) sal[row] = red[0] + b2[0];
}

// ---------------------------------------------------------------------------
// Deterministic top-k via rank counting (stable tie-break by lower index,
// matching jax.lax.top_k). One block per batch.
// ---------------------------------------------------------------------------
__global__ __launch_bounds__(256)
void topk_mask_kernel(const float* __restrict__ sal, int* __restrict__ mask) {
    __shared__ float s[N_];
    const int b = blockIdx.x, tid = threadIdx.x;
    for (int i = tid; i < N_; i += 256) s[i] = sal[b * N_ + i];
    __syncthreads();
    for (int n = tid; n < N_; n += 256) {
        const float v = s[n];
        int rank = 0;
        for (int j = 0; j < N_; ++j) {
            const float u = s[j];
            rank += (u > v) || (u == v && j < n);
        }
        mask[b * N_ + n] = (rank < TOPK_) ? 1 : 0;
    }
}

// ---------------------------------------------------------------------------
// fp32 -> bf16 (elementwise) and fp32 [K,N] -> bf16 [N,K] transpose-convert.
// ---------------------------------------------------------------------------
__global__ __launch_bounds__(256)
void cvt_bf16_kernel(const float* __restrict__ in, __bf16* __restrict__ out, size_t n) {
    size_t i = (size_t)blockIdx.x * 256 + threadIdx.x;
    if (i < n) out[i] = (__bf16)in[i];
}

__global__ __launch_bounds__(256)
void cvt_transpose_bf16_kernel(const float* __restrict__ in, __bf16* __restrict__ out,
                               int K, int N) {   // in[K][N] -> out[N][K]
    size_t i = (size_t)blockIdx.x * 256 + threadIdx.x;
    if (i < (size_t)K * N) {
        int n = (int)(i / K);
        int k = (int)(i % K);
        out[i] = (__bf16)in[(size_t)k * N + n];
    }
}

// ---------------------------------------------------------------------------
// Generic batched bf16 WMMA GEMM:  C[z] = scale * A[z] @ B[z] + bias
//   block tile 128(M) x 64(N), K-step 32, 8 waves of 32x32 wave-tiles.
//   AF32   : A is fp32 in memory, converted to bf16 while staging
//   TRANSB : B stored [N][K] row-major (contiguous staging); else [K][N] gather
//   OUTF32 : write fp32 (Cf) else bf16 (Cb)
//   batch z -> (b = z/Hdim, h = z%Hdim), offsets = b*StrB + h*StrH (elements).
// VGPR lane layouts follow CDNA5 ISA 7.12.2 (wave32):
//   A 16x32 bf16: lane L (g=L>>4): elems 0..7 -> K=g*8+i ; 8..15 -> K=16+g*8+(i-8)
//   B 32x16 bf16: lane L: elems 0..7 -> K=g*8+i ; 8..15 -> K=16+g*8+(i-8), col = L&15
//   C 16x16 f32 : lane L: vgpr r -> M = g*8+r, col = L&15
// ---------------------------------------------------------------------------
template <int AF32, int TRANSB, int OUTF32>
__global__ __launch_bounds__(256)
void wmma_gemm_kernel(const void* __restrict__ Av,
                      const __bf16* __restrict__ Bm,
                      int M, int Ncols, int K,
                      int lda, int ldb, int ldc, int Hdim,
                      long aStrB, long aStrH, long bStrB, long bStrH,
                      long cStrB, long cStrH,
                      const float* __restrict__ bias, float scale,
                      float* __restrict__ Cf, __bf16* __restrict__ Cb) {
    __shared__ __align__(16) __bf16 As[128][40];   // 32 cols + 8 pad (bank skew)
    __shared__ __align__(16) __bf16 Bs[64][40];    // stored n-major: Bs[n][k]

    const int tid  = threadIdx.x;
    const int lane = tid & 31;
    const int wave = tid >> 5;
    const int g    = lane >> 4;       // half-wave group
    const int lm   = lane & 15;

    const int z  = blockIdx.z;
    const int bb = z / Hdim;
    const int hh = z % Hdim;
    const long aOff = (long)bb * aStrB + (long)hh * aStrH;
    const long bOff = (long)bb * bStrB + (long)hh * bStrH;
    const long cOff = (long)bb * cStrB + (long)hh * cStrH;

    const __bf16* Abf = (const __bf16*)Av;
    const float*  Af  = (const float*)Av;

    const int blockM = blockIdx.y * 128;
    const int blockN = blockIdx.x * 64;
    const int waveM  = (wave & 3) * 32;
    const int waveN  = (wave >> 2) * 32;

    v8f acc[2][2];
    #pragma unroll
    for (int i = 0; i < 2; ++i)
        #pragma unroll
        for (int j = 0; j < 2; ++j)
            #pragma unroll
            for (int e = 0; e < 8; ++e) acc[i][j][e] = 0.0f;

    for (int kk0 = 0; kk0 < K; kk0 += 32) {
        __syncthreads();
        // ---- stage A tile 128x32 into LDS ----
        if (AF32) {
            #pragma unroll
            for (int c0 = 0; c0 < 128 * 32 / 4; c0 += 256) {     // 4 floats/chunk
                const int c = c0 + tid;
                const int r = c >> 3;
                const int s = (c & 7) * 4;
                const float4 f = *(const float4*)(Af + aOff + (long)(blockM + r) * lda + kk0 + s);
                As[r][s + 0] = (__bf16)f.x; As[r][s + 1] = (__bf16)f.y;
                As[r][s + 2] = (__bf16)f.z; As[r][s + 3] = (__bf16)f.w;
            }
        } else {
            #pragma unroll
            for (int c0 = 0; c0 < 128 * 32 / 8; c0 += 256) {     // 8 bf16/chunk
                const int c = c0 + tid;
                const int r = c >> 2;
                const int s = (c & 3) * 8;
                stage16(Abf + aOff + (long)(blockM + r) * lda + kk0 + s, &As[r][s]);
            }
        }
        // ---- stage B tile 32x64 into LDS (n-major) ----
        if (TRANSB) {
            {                                                    // 256 chunks total
                const int c = tid;
                const int r = c >> 2;
                const int s = (c & 3) * 8;
                stage16(Bm + bOff + (long)(blockN + r) * ldb + kk0 + s, &Bs[r][s]);
            }
        } else {
            #pragma unroll
            for (int e0 = 0; e0 < 64 * 32; e0 += 256) {
                const int e  = e0 + tid;
                const int n  = e & 63;
                const int kk = e >> 6;
                Bs[n][kk] = Bm[bOff + (long)(kk0 + kk) * ldb + blockN + n];
            }
        }
        // ---- prefetch next K tile (global_prefetch_b8) ----
        if (kk0 + 32 < K) {
            const char* pa = AF32
                ? (const char*)(Af  + aOff + (long)(blockM + (tid >> 1)) * lda + kk0 + 32)
                : (const char*)(Abf + aOff + (long)(blockM + (tid >> 1)) * lda + kk0 + 32);
            __builtin_prefetch(pa, 0, 0);
        }
        stage_wait();                   // drain ASYNCcnt before the barrier
        __syncthreads();

        // ---- 2x2 wave tiles, one 16x16x32 WMMA each ----
        #pragma unroll
        for (int mi = 0; mi < 2; ++mi) {
            const int m = waveM + mi * 16 + lm;
            const v8bf alo = *(const v8bf*)&As[m][g * 8];
            const v8bf ahi = *(const v8bf*)&As[m][16 + g * 8];
            v16bf afrag;
            #pragma unroll
            for (int i = 0; i < 8; ++i) { afrag[i] = alo[i]; afrag[i + 8] = ahi[i]; }
            #pragma unroll
            for (int ni = 0; ni < 2; ++ni) {
                const int n = waveN + ni * 16 + lm;
                const v8bf b0 = *(const v8bf*)&Bs[n][g * 8];
                const v8bf b1 = *(const v8bf*)&Bs[n][16 + g * 8];
                v16bf bfrag;
                #pragma unroll
                for (int i = 0; i < 8; ++i) { bfrag[i] = b0[i]; bfrag[i + 8] = b1[i]; }
                acc[mi][ni] = __builtin_amdgcn_wmma_f32_16x16x32_bf16(
                    /*neg_a=*/false, afrag, /*neg_b=*/false, bfrag,
                    /*c_mod=*/(short)0, acc[mi][ni],
                    /*reuse_a=*/false, /*reuse_b=*/false);
            }
        }
    }

    // ---- epilogue: scale, bias, store fp32 or bf16 ----
    #pragma unroll
    for (int mi = 0; mi < 2; ++mi) {
        #pragma unroll
        for (int ni = 0; ni < 2; ++ni) {
            const int col = blockN + waveN + ni * 16 + lm;
            const float bv = bias ? bias[col] : 0.0f;
            #pragma unroll
            for (int r = 0; r < 8; ++r) {
                const int m = blockM + waveM + mi * 16 + g * 8 + r;
                const float val = acc[mi][ni][r] * scale + bv;
                if (OUTF32) Cf[cOff + (long)m * ldc + col] = val;
                else        Cb[cOff + (long)m * ldc + col] = (__bf16)val;
            }
        }
    }
}

// ---------------------------------------------------------------------------
// Masked row softmax, in place on the attn slab. One block per (b,h,q) row.
// Masked-out columns behave as -inf -> exactly 0 after softmax.
// ---------------------------------------------------------------------------
__global__ __launch_bounds__(256)
void softmax_kernel(float* __restrict__ attn, const int* __restrict__ mask) {
    __shared__ float red[256];
    const size_t row = blockIdx.x;                 // b*H*N + h*N + q
    const int b = (int)(row / ((size_t)H_ * N_));
    float* p = attn + row * (size_t)N_;
    const int* mrow = mask + b * N_;
    const int tid = threadIdx.x;

    float v[8];
    float mx = -3.4e38f;
    #pragma unroll
    for (int i = 0; i < 8; ++i) {
        const int j = tid + i * 256;
        const float val = mrow[j] ? p[j] : -3.4e38f;
        v[i] = val;
        mx = fmaxf(mx, val);
    }
    red[tid] = mx; __syncthreads();
    for (int s = 128; s > 0; s >>= 1) {
        if (tid < s) red[tid] = fmaxf(red[tid], red[tid + s]);
        __syncthreads();
    }
    const float M = red[0]; __syncthreads();

    float sum = 0.0f;
    #pragma unroll
    for (int i = 0; i < 8; ++i) {
        const float e = (v[i] <= -3.0e38f) ? 0.0f : __expf(v[i] - M);
        v[i] = e; sum += e;
    }
    red[tid] = sum; __syncthreads();
    for (int s = 128; s > 0; s >>= 1) {
        if (tid < s) red[tid] += red[tid + s];
        __syncthreads();
    }
    const float inv = 1.0f / red[0];
    #pragma unroll
    for (int i = 0; i < 8; ++i) p[tid + i * 256] = v[i] * inv;
}

// ---------------------------------------------------------------------------
extern "C" void kernel_launch(void* const* d_in, const int* in_sizes, int n_in,
                              void* d_out, int out_size, void* d_ws, size_t ws_size,
                              hipStream_t stream) {
    (void)in_sizes; (void)n_in; (void)out_size; (void)ws_size;

    const float* x      = (const float*)d_in[0];   // [B,N,C]
    const float* qkv_w  = (const float*)d_in[1];   // [C,3C]
    const float* qkv_b  = (const float*)d_in[2];   // [3C]
    const float* proj_w = (const float*)d_in[3];   // [C,C]
    const float* proj_b = (const float*)d_in[4];   // [C]
    const float* sal_w1 = (const float*)d_in[5];   // [C,C/2]
    const float* sal_b1 = (const float*)d_in[6];   // [C/2]
    const float* sal_w2 = (const float*)d_in[7];   // [C/2,1]
    const float* sal_b2 = (const float*)d_in[8];   // [1]

    float* out  = (float*)d_out;                   // [B,N,C]
    float* attn = out + (size_t)B_ * N_ * C_;      // [B,H,N,N]

    // workspace carve-out (256B aligned)
    char* ws = (char*)d_ws;
    auto carve = [&ws](size_t bytes) {
        char* p = ws;
        ws += (bytes + 255) & ~(size_t)255;
        return p;
    };
    __bf16* xbf    = (__bf16*)carve((size_t)B_ * N_ * C_  * 2);  // x in bf16
    __bf16* wqkvT  = (__bf16*)carve((size_t)K3_ * C_      * 2);  // qkv_w^T
    __bf16* wprojT = (__bf16*)carve((size_t)C_ * C_       * 2);  // proj_w^T
    __bf16* qkv    = (__bf16*)carve((size_t)B_ * N_ * K3_ * 2);  // [B*N, 3C]
    __bf16* otmp   = (__bf16*)carve((size_t)B_ * N_ * C_  * 2);  // heads merged
    float*  sal    = (float*) carve((size_t)B_ * N_ * 4);
    int*    mask   = (int*)   carve((size_t)B_ * N_ * 4);

    // 1) saliency + top-k mask
    sal_kernel<<<B_ * N_, CH_, 0, stream>>>(x, sal_w1, sal_b1, sal_w2, sal_b2, sal);
    topk_mask_kernel<<<B_, 256, 0, stream>>>(sal, mask);

    // 2) bf16 conversions (weights stored transposed -> contiguous B staging)
    {
        size_t n = (size_t)B_ * N_ * C_;
        cvt_bf16_kernel<<<(unsigned)((n + 255) / 256), 256, 0, stream>>>(x, xbf, n);
        size_t t1 = (size_t)C_ * K3_;
        cvt_transpose_bf16_kernel<<<(unsigned)((t1 + 255) / 256), 256, 0, stream>>>(
            qkv_w, wqkvT, C_, K3_);
        size_t t2 = (size_t)C_ * C_;
        cvt_transpose_bf16_kernel<<<(unsigned)((t2 + 255) / 256), 256, 0, stream>>>(
            proj_w, wprojT, C_, C_);
    }

    // 3) qkv = x @ qkv_w + qkv_b  (bf16 out, [B*N, 2304])
    wmma_gemm_kernel<0, 1, 0><<<dim3(K3_ / 64, (B_ * N_) / 128, 1), 256, 0, stream>>>(
        xbf, wqkvT,
        B_ * N_, K3_, C_, C_, C_, K3_, /*Hdim=*/1,
        0, 0, 0, 0, 0, 0, qkv_b, 1.0f, nullptr, qkv);

    // 4) S = (q @ k^T) * hd^-0.5   -> attn slab (fp32), per (b,h)
    wmma_gemm_kernel<0, 1, 1><<<dim3(N_ / 64, N_ / 128, B_ * H_), 256, 0, stream>>>(
        qkv, qkv + 768,
        N_, N_, HD_, K3_, K3_, N_, /*Hdim=*/H_,
        (long)N_ * K3_, (long)HD_,                  // A (q) strides
        (long)N_ * K3_, (long)HD_,                  // B (k) strides
        (long)H_ * N_ * N_, (long)N_ * N_,          // C (attn) strides
        nullptr, 0.125f, attn, nullptr);

    // 5) masked softmax in place on attn
    softmax_kernel<<<B_ * H_ * N_, 256, 0, stream>>>(attn, mask);

    // 6) head_out = attn @ v  (A fp32->bf16 on stage; bf16 out merged [B,N,C])
    wmma_gemm_kernel<1, 0, 0><<<dim3(HD_ / 64, N_ / 128, B_ * H_), 256, 0, stream>>>(
        attn, qkv + 1536,
        N_, HD_, N_, N_, K3_, C_, /*Hdim=*/H_,
        (long)H_ * N_ * N_, (long)N_ * N_,          // A (attn) strides
        (long)N_ * K3_, (long)HD_,                  // B (v) strides
        (long)N_ * C_, (long)HD_,                   // C (otmp) strides (h*64 col off)
        nullptr, 1.0f, nullptr, otmp);

    // 7) out = head_out @ proj_w + proj_b  (fp32 out)
    wmma_gemm_kernel<0, 1, 1><<<dim3(C_ / 64, (B_ * N_) / 128, 1), 256, 0, stream>>>(
        otmp, wprojT,
        B_ * N_, C_, C_, C_, C_, C_, /*Hdim=*/1,
        0, 0, 0, 0, 0, 0, proj_b, 1.0f, out, nullptr);
}
